// _PropagateBodyWrapper_42898133352843
// MI455X (gfx1250) — matrix operand order repaired and checked by reading, compiled, not run on hardware
//
#include <hip/hip_runtime.h>

// ---------------------------------------------------------------------------
// Problem constants (from the reference)
// ---------------------------------------------------------------------------
constexpr int Bn  = 2;
constexpr int Cc  = 64;
constexpr int Hh  = 96;
constexpr int Ww  = 96;
constexpr int HW  = Hh * Ww;        // 9216
constexpr int P   = Bn * HW;        // 18432 output pixels
constexpr int DG  = 16;             // deform groups
constexpr int KT  = 9;              // 3x3 taps
constexpr int CPG = 8;              // channels per deform group
constexpr int KKD = DG * CPG * KT;  // 1152 (deform GEMM K, multiple of 32)
constexpr int COFF = 27 * DG;       // 432
constexpr int CIN0 = 3 * Cc + 4;    // 196
constexpr int CIN0P = 224;          // 196 padded to multiple of 32 (NHWC stride)
constexpr int WPB   = 8;            // waves per block (256 threads, wave32)

typedef _Float16 half_t;
typedef __attribute__((ext_vector_type(16))) _Float16 v16h;
typedef __attribute__((ext_vector_type(8)))  _Float16 v8h;
typedef __attribute__((ext_vector_type(8)))  float    v8f;

#define DEV static __device__ __forceinline__

union AFrag { v16h v; v8h h[2]; };

// ---------------------------------------------------------------------------
// CDNA5 f16 WMMA fragment layouts (CDNA5 ISA) are K-contiguous per lane:
//   A 16x32: lane L (m=L&15) holds K = [k0+off .. +7] ++ [k0+16+off .. +7],
//            off = 8*(L>=16)  -> two 16B vector loads from a K-major row.
//   B 32x16: lane L (n=L&15) holds K = [k0+khi .. +7] ++ [k0+16+khi .. +7], khi = 8*(L>=16)
//   C/D f32: lane L (n=L&15), elem f -> m = f + 8*(L>=16)
// ---------------------------------------------------------------------------
DEV v16h frag_from_row(const half_t* __restrict__ row, int k0, int off) {
  AFrag u;
  u.h[0] = *(const v8h*)(row + k0 + off);
  u.h[1] = *(const v8h*)(row + k0 + 16 + off);
  return u.v;
}
DEV v16h frag_b_from_lds(const half_t* row, int k0, int khi) {
  AFrag u;
  u.h[0] = *(const v8h*)(row + k0 + khi);
  u.h[1] = *(const v8h*)(row + k0 + khi + 16);
  return u.v;
}

// ---------------------------------------------------------------------------
// Elementwise helpers
// ---------------------------------------------------------------------------
__global__ void k_f32_to_f16(const float* __restrict__ s, half_t* __restrict__ d, int n) {
  int i = blockIdx.x * blockDim.x + threadIdx.x;
  if (i < n) d[i] = (half_t)s[i];
}

__global__ void k_zero16(half_t* __restrict__ d, int n) {
  int i = blockIdx.x * blockDim.x + threadIdx.x;
  if (i < n) d[i] = (half_t)0.f;
}

// Repack OIHW f32 -> [tap][Cout][CinPad] f16 (zero pad channels)
__global__ void k_repack_w(const float* __restrict__ s, half_t* __restrict__ d,
                           int Cin, int CinPad, int Cout) {
  int i = blockIdx.x * blockDim.x + threadIdx.x;
  int n = 9 * Cout * CinPad;
  if (i >= n) return;
  int c   = i % CinPad;
  int o   = (i / CinPad) % Cout;
  int tap = i / (CinPad * Cout);
  d[i] = (c < Cin) ? (half_t)s[((size_t)o * Cin + c) * 9 + tap] : (half_t)0.f;
}

// ---------------------------------------------------------------------------
// grid_sample warps + concat -> NHWC f16 [P, 224] (pad channels zeroed)
// ---------------------------------------------------------------------------
DEV void sample64(const float* __restrict__ img, const float* __restrict__ grid,
                  int b, int r, half_t* __restrict__ dstRow, int cbase) {
  float gx = grid[(size_t)(b * HW + r) * 2 + 0];
  float gy = grid[(size_t)(b * HW + r) * 2 + 1];
  float xf = (gx + 1.f) * 0.5f * (Ww - 1);
  float yf = (gy + 1.f) * 0.5f * (Hh - 1);
  float x0f = floorf(xf), y0f = floorf(yf);
  int ix = (int)x0f, iy = (int)y0f;
  float wx1 = xf - x0f, wx0 = 1.f - wx1;
  float wy1 = yf - y0f, wy0 = 1.f - wy1;
  bool vx0 = (ix >= 0) && (ix < Ww), vx1 = (ix + 1 >= 0) && (ix + 1 < Ww);
  bool vy0 = (iy >= 0) && (iy < Hh), vy1 = (iy + 1 >= 0) && (iy + 1 < Hh);
  int cx0 = min(max(ix, 0), Ww - 1), cx1 = min(max(ix + 1, 0), Ww - 1);
  int cy0 = min(max(iy, 0), Hh - 1), cy1 = min(max(iy + 1, 0), Hh - 1);
  float w00 = (vy0 && vx0) ? wy0 * wx0 : 0.f;
  float w01 = (vy0 && vx1) ? wy0 * wx1 : 0.f;
  float w10 = (vy1 && vx0) ? wy1 * wx0 : 0.f;
  float w11 = (vy1 && vx1) ? wy1 * wx1 : 0.f;
  const float* base = img + (size_t)b * Cc * HW;
  int i00 = cy0 * Ww + cx0, i01 = cy0 * Ww + cx1;
  int i10 = cy1 * Ww + cx0, i11 = cy1 * Ww + cx1;
  for (int c = 0; c < Cc; ++c) {
    const float* ch = base + (size_t)c * HW;
    float acc = w00 * ch[i00] + w01 * ch[i01] + w10 * ch[i10] + w11 * ch[i11];
    dstRow[cbase + c] = (half_t)acc;
  }
}

__global__ void k_prep_xcat(const float* __restrict__ fp,  const float* __restrict__ g1,
                            const float* __restrict__ fn2, const float* __restrict__ g2,
                            const float* __restrict__ fc,  const float* __restrict__ fl1,
                            const float* __restrict__ fl2, half_t* __restrict__ xcat) {
  int p = blockIdx.x * blockDim.x + threadIdx.x;
  if (p >= P) return;
  int b = p / HW, r = p % HW;
  half_t* row = xcat + (size_t)p * CIN0P;
  sample64(fp,  g1, b, r, row, 0);
  for (int c = 0; c < Cc; ++c)
    row[64 + c] = (half_t)fc[((size_t)(b * Cc + c) * HW) + r];
  sample64(fn2, g2, b, r, row, 128);
  row[192] = (half_t)fl1[((size_t)(b * 2 + 0) * HW) + r];
  row[193] = (half_t)fl1[((size_t)(b * 2 + 1) * HW) + r];
  row[194] = (half_t)fl2[((size_t)(b * 2 + 0) * HW) + r];
  row[195] = (half_t)fl2[((size_t)(b * 2 + 1) * HW) + r];
  for (int c = CIN0; c < CIN0P; ++c) row[c] = (half_t)0.f;   // pad
}

// ---------------------------------------------------------------------------
// 3x3 conv (stride1,pad1) = 9 tap-GEMMs on WMMA.
//   Activations: NHWC f16 [P, CinPad]. Weights: [tap][Cout][CinPad] f16.
//   Block = 8 waves sharing one cout-tile; weight tile [9][16][CinPad] is
//   cooperatively staged into dynamic LDS once (ds_load_b128 B fragments),
//   each wave computes a different 16-pixel tile. OOB taps redirect the A-row
//   pointer to a shared zero row (branch-free inner loop).
//   act: 0 none, 1 lrelu(0.1), 2 relu.
// ---------------------------------------------------------------------------
__global__ void k_conv3x3_wmma(const half_t* __restrict__ in16, int CinPad,
                               const half_t* __restrict__ wT,
                               const half_t* __restrict__ zrow,
                               const float*  __restrict__ bias, int Cout,
                               const float*  __restrict__ res,     // (B,Cout,H,W) optional
                               float*  __restrict__ out32,         // NCHW, optional
                               half_t* __restrict__ out16,         // NHWC, optional
                               int c16tot, int c16base, int act) {
  extern __shared__ half_t ldsW[];                 // [9][16][CinPad]
  const int numCTiles = Cout / 16;
  const int cTile     = blockIdx.x % numCTiles;
  const int pixGroup  = blockIdx.x / numCTiles;
  const int waveId    = threadIdx.x >> 5;
  const int lane      = threadIdx.x & 31;
  const int coutBase  = cTile * 16;

  // ---- cooperative weight staging into LDS (v8h granules) ---------------
  {
    const int chunks = (9 * 16 * CinPad) / 8;
    for (int chk = threadIdx.x; chk < chunks; chk += blockDim.x) {
      int idx  = chk * 8;                 // half index in [tapR][CinPad]
      int tapR = idx / CinPad;            // tap*16 + r
      int kk   = idx - tapR * CinPad;
      int tap  = tapR >> 4, r = tapR & 15;
      ((v8h*)ldsW)[chk] =
          *(const v8h*)(wT + ((size_t)tap * Cout + coutBase + r) * CinPad + kk);
    }
  }
  __syncthreads();

  const int pixTile = pixGroup * WPB + waveId;     // exact grid: always valid
  const int pixBase = pixTile * 16;
  const int b       = pixBase / HW;
  const int rem     = pixBase % HW;
  const int y       = rem / Ww;
  const int x0      = rem % Ww;

  const int m    = lane & 15;
  const int n    = lane & 15;
  const int offA = (lane >> 4) << 3;    // 0 / 8
  const int khiB = (lane >> 4) << 3;    // 0 / 8

  v8f acc = {};
#pragma unroll
  for (int tap = 0; tap < 9; ++tap) {
    const int dy = tap / 3 - 1, dx = tap % 3 - 1;
    const int yy = y + dy;                // uniform across wave
    const int xx = x0 + m + dx;           // per lane
    const bool valid = (yy >= 0) && (yy < Hh) && (xx >= 0) && (xx < Ww);
    const half_t* arow = valid ? (in16 + ((size_t)(b * Hh + yy) * Ww + xx) * CinPad)
                               : zrow;    // branch-free: cndmask on address
    const half_t* wrow = ldsW + ((tap << 4) + n) * CinPad;
    for (int k0 = 0; k0 < CinPad; k0 += 32) {
      v16h a  = frag_from_row(arow, k0, offA);
      v16h bm = frag_b_from_lds(wrow, k0, khiB);
      acc = __builtin_amdgcn_wmma_f32_16x16x32_f16(false, a, false, bm,
                                                   (short)0, acc, false, false);
    }
  }

  const int co = coutBase + n;
  const float bs = bias[co];
  const int mAdd = (lane >> 4) << 3;
#pragma unroll
  for (int f = 0; f < 8; ++f) {
    int mm = f + mAdd;
    size_t oidx = ((size_t)(b * Cout + co) * HW) + rem + mm;
    float v = acc[f] + bs;
    if (res) v += res[oidx];
    if (act == 1)      v = (v > 0.f) ? v : 0.1f * v;
    else if (act == 2) v = (v > 0.f) ? v : 0.f;
    if (out32) out32[oidx] = v;
    if (out16) out16[(size_t)(pixBase + mm) * c16tot + c16base + co] = (half_t)v;
  }
}

// ---------------------------------------------------------------------------
// offset / mask post-processing of the offset-head output (B,432,H,W)
// ---------------------------------------------------------------------------
__global__ void k_offset_mask(const float* __restrict__ co3,
                              const float* __restrict__ fl1, const float* __restrict__ fl2,
                              float* __restrict__ offs, float* __restrict__ msk) {
  int i = blockIdx.x * blockDim.x + threadIdx.x;
  if (i >= Bn * COFF * HW) return;
  int b  = i / (COFF * HW);
  int r0 = i % (COFF * HW);
  int ch = r0 / HW;
  int p  = r0 % HW;
  float v = co3[i];
  if (ch < 288) {
    float o = 10.f * tanhf(v);
    const float* fl = (ch < 144) ? fl1 : fl2;
    int j = (ch < 144) ? ch : (ch - 144);
    o += fl[((size_t)(b * 2 + (1 - (j & 1))) * HW) + p];   // flow[:, ::-1] tiled
    offs[((size_t)(b * 288 + ch) * HW) + p] = o;
  } else {
    msk[((size_t)(b * 144 + (ch - 288)) * HW) + p] = 1.f / (1.f + expf(-v));
  }
}

// ---------------------------------------------------------------------------
// modulated deformable sampling -> dense f16 A matrix [P, 1152]
//   kk = (g*CPG + c)*9 + k   (matches flat dc_weight [Cout, 128, 3, 3])
// ---------------------------------------------------------------------------
__global__ void k_deform_val(const float* __restrict__ fp, const float* __restrict__ fn2,
                             const float* __restrict__ offs, const float* __restrict__ msk,
                             half_t* __restrict__ val16) {
  int i = blockIdx.x * blockDim.x + threadIdx.x;
  if (i >= Bn * HW * DG * KT) return;
  int k = i % KT;  int t = i / KT;
  int g = t % DG;  t /= DG;
  int p = t % HW;  int b = t / HW;
  int y = p / Ww, x = p % Ww;

  float dy = offs[((size_t)(b * 288 + (g * KT + k) * 2 + 0) * HW) + p];
  float dx = offs[((size_t)(b * 288 + (g * KT + k) * 2 + 1) * HW) + p];
  float m  = msk [((size_t)(b * 144 +  g * KT + k)          * HW) + p];

  float py = dy + (float)(k / 3) + (float)y - 1.f;
  float px = dx + (float)(k % 3) + (float)x - 1.f;
  float y0f = floorf(py), x0f = floorf(px);
  int iy = (int)y0f, ix = (int)x0f;
  float wx1 = px - x0f, wx0 = 1.f - wx1;
  float wy1 = py - y0f, wy0 = 1.f - wy1;
  bool vx0 = (ix >= 0) && (ix < Ww), vx1 = (ix + 1 >= 0) && (ix + 1 < Ww);
  bool vy0 = (iy >= 0) && (iy < Hh), vy1 = (iy + 1 >= 0) && (iy + 1 < Hh);
  int cx0 = min(max(ix, 0), Ww - 1), cx1 = min(max(ix + 1, 0), Ww - 1);
  int cy0 = min(max(iy, 0), Hh - 1), cy1 = min(max(iy + 1, 0), Hh - 1);
  float w00 = (vy0 && vx0) ? wy0 * wx0 : 0.f;
  float w01 = (vy0 && vx1) ? wy0 * wx1 : 0.f;
  float w10 = (vy1 && vx0) ? wy1 * wx0 : 0.f;
  float w11 = (vy1 && vx1) ? wy1 * wx1 : 0.f;
  int i00 = cy0 * Ww + cx0, i01 = cy0 * Ww + cx1;
  int i10 = cy1 * Ww + cx0, i11 = cy1 * Ww + cx1;

  size_t rowBase = (size_t)(b * HW + p) * KKD + (size_t)g * (CPG * KT) + k;
  for (int c = 0; c < CPG; ++c) {
    int cg = g * CPG + c;
    const float* ch = (cg < Cc) ? (fp  + (size_t)(b * Cc + cg)      * HW)
                                : (fn2 + (size_t)(b * Cc + cg - Cc) * HW);
    float acc = w00 * ch[i00] + w01 * ch[i01] + w10 * ch[i10] + w11 * ch[i11];
    val16[rowBase + (size_t)c * KT] = (half_t)(acc * m);
  }
}

// ---------------------------------------------------------------------------
// dense WMMA GEMM for the deform einsum: [P,1152] x [1152,64]
//   Block = 8 waves sharing one cout-tile; the 16x1152 weight tile (36 KB) is
//   staged into dynamic LDS once and B fragments come from ds_load_b128.
// ---------------------------------------------------------------------------
__global__ void k_dgemm_wmma(const half_t* __restrict__ A,    // [P, KKD]
                             const half_t* __restrict__ w16,  // [64, KKD]
                             const float*  __restrict__ bias,
                             float*  __restrict__ out32,              // NCHW
                             half_t* __restrict__ out16, int c16tot, int c16base) {
  extern __shared__ half_t ldsW[];                 // [16][KKD]
  const int numCTiles = Cc / 16;                   // 4
  const int cTile     = blockIdx.x % numCTiles;
  const int pixGroup  = blockIdx.x / numCTiles;
  const int waveId    = threadIdx.x >> 5;
  const int lane      = threadIdx.x & 31;
  const int coutBase  = cTile * 16;

  {
    const int chunks = (16 * KKD) / 8;             // 2304
    for (int chk = threadIdx.x; chk < chunks; chk += blockDim.x) {
      int idx = chk * 8;
      int r   = idx / KKD;
      int kk  = idx - r * KKD;
      ((v8h*)ldsW)[chk] = *(const v8h*)(w16 + (size_t)(coutBase + r) * KKD + kk);
    }
  }
  __syncthreads();

  const int pixTile = pixGroup * WPB + waveId;
  const int pixBase = pixTile * 16;
  const int b   = pixBase / HW;
  const int rem = pixBase % HW;

  const int m    = lane & 15, n = lane & 15;
  const int offA = (lane >> 4) << 3;
  const int khiB = (lane >> 4) << 3;
  const half_t* arow = A + (size_t)(pixBase + m) * KKD;
  const half_t* wrow = ldsW + n * KKD;

  v8f acc = {};
#pragma unroll 4
  for (int k0 = 0; k0 < KKD; k0 += 32) {
    v16h a  = frag_from_row(arow, k0, offA);
    v16h bm = frag_b_from_lds(wrow, k0, khiB);
    acc = __builtin_amdgcn_wmma_f32_16x16x32_f16(false, a, false, bm,
                                                 (short)0, acc, false, false);
  }

  const int co = coutBase + n;
  const float bs = bias[co];
  const int mAdd = (lane >> 4) << 3;
#pragma unroll
  for (int f = 0; f < 8; ++f) {
    int mm = f + mAdd;
    float v = acc[f] + bs;
    out32[((size_t)(b * Cc + co) * HW) + rem + mm] = v;
    out16[(size_t)(pixBase + mm) * c16tot + c16base + co] = (half_t)v;
  }
}

// ---------------------------------------------------------------------------
// backbone_prefix (B,128,H,W) f32 -> featcat16 NHWC channels 0..127
// ---------------------------------------------------------------------------
__global__ void k_prefix16(const float* __restrict__ s, half_t* __restrict__ fcat) {
  int i = blockIdx.x * blockDim.x + threadIdx.x;
  if (i >= Bn * 2 * Cc * HW) return;
  int b = i / (2 * Cc * HW);
  int r = i % (2 * Cc * HW);
  int c = r / HW, p = r % HW;
  fcat[(size_t)(b * HW + p) * (3 * Cc) + c] = (half_t)s[i];
}

__global__ void k_final_add(const float* __restrict__ a, const float* __restrict__ b,
                            float* __restrict__ o, int n) {
  int i = blockIdx.x * blockDim.x + threadIdx.x;
  if (i < n) o[i] = a[i] + b[i];
}

// ---------------------------------------------------------------------------
// Host launcher
// ---------------------------------------------------------------------------
extern "C" void kernel_launch(void* const* d_in, const int* in_sizes, int n_in,
                              void* d_out, int out_size, void* d_ws, size_t ws_size,
                              hipStream_t stream) {
  const float* feat_prop = (const float*)d_in[0];
  const float* grid_n1   = (const float*)d_in[1];
  const float* feat_n2   = (const float*)d_in[2];
  const float* grid_n2   = (const float*)d_in[3];
  const float* feat_cur  = (const float*)d_in[4];
  const float* flow_1    = (const float*)d_in[5];
  const float* flow_2    = (const float*)d_in[6];
  const float* bb_prefix = (const float*)d_in[7];
  const float* co_w0 = (const float*)d_in[8];   const float* co_b0 = (const float*)d_in[9];
  const float* co_w1 = (const float*)d_in[10];  const float* co_b1 = (const float*)d_in[11];
  const float* co_w2 = (const float*)d_in[12];  const float* co_b2 = (const float*)d_in[13];
  const float* co_w3 = (const float*)d_in[14];  const float* co_b3 = (const float*)d_in[15];
  const float* dc_w  = (const float*)d_in[16];  const float* dc_b  = (const float*)d_in[17];
  const float* bb_w0 = (const float*)d_in[18];  const float* bb_b0 = (const float*)d_in[19];
  const float* r1w1  = (const float*)d_in[20];  const float* r1b1  = (const float*)d_in[21];
  const float* r1w2  = (const float*)d_in[22];  const float* r1b2  = (const float*)d_in[23];
  const float* r2w1  = (const float*)d_in[24];  const float* r2b1  = (const float*)d_in[25];
  const float* r2w2  = (const float*)d_in[26];  const float* r2b2  = (const float*)d_in[27];

  // ---- workspace carve-out (256B aligned) ------------------------------
  char* base = (char*)d_ws;
  size_t off = 0;
  auto carve = [&](size_t bytes) -> char* {
    char* p = base + off;
    off = (off + bytes + 255) & ~(size_t)255;
    return p;
  };
  half_t* w0_16  = (half_t*)carve((size_t)9 * Cc * CIN0P * 2);     // [tap][64][224]
  half_t* w1_16  = (half_t*)carve((size_t)9 * Cc * Cc * 2);
  half_t* w2_16  = (half_t*)carve((size_t)9 * Cc * Cc * 2);
  half_t* w3_16  = (half_t*)carve((size_t)9 * COFF * Cc * 2);
  half_t* wdc16  = (half_t*)carve((size_t)Cc * KKD * 2);           // flat [64][1152]
  half_t* wbb16  = (half_t*)carve((size_t)9 * Cc * 3 * Cc * 2);
  half_t* wr1a16 = (half_t*)carve((size_t)9 * Cc * Cc * 2);
  half_t* wr1b16 = (half_t*)carve((size_t)9 * Cc * Cc * 2);
  half_t* wr2a16 = (half_t*)carve((size_t)9 * Cc * Cc * 2);
  half_t* wr2b16 = (half_t*)carve((size_t)9 * Cc * Cc * 2);
  half_t* zrow   = (half_t*)carve((size_t)CIN0P * 2);              // shared zero row
  half_t* xcat16 = (half_t*)carve((size_t)P * CIN0P * 2);          // NHWC
  half_t* hA16   = (half_t*)carve((size_t)P * Cc * 2);
  half_t* hB16   = (half_t*)carve((size_t)P * Cc * 2);
  float*  co3o32 = (float*) carve((size_t)P * COFF * 4);
  float*  off32  = (float*) carve((size_t)P * 288 * 4);
  float*  msk32  = (float*) carve((size_t)P * 144 * 4);
  half_t* val16  = (half_t*)carve((size_t)P * KKD * 2);
  float*  fpn32  = (float*) carve((size_t)P * Cc * 4);
  half_t* fcat16 = (half_t*)carve((size_t)P * 3 * Cc * 2);
  float*  h0_32  = (float*) carve((size_t)P * Cc * 4);
  half_t* h0_16  = (half_t*)carve((size_t)P * Cc * 2);
  half_t* rA16   = (half_t*)carve((size_t)P * Cc * 2);
  float*  h1_32  = (float*) carve((size_t)P * Cc * 4);
  half_t* h1_16  = (half_t*)carve((size_t)P * Cc * 2);
  half_t* rB16   = (half_t*)carve((size_t)P * Cc * 2);
  float*  h2_32  = (float*) carve((size_t)P * Cc * 4);
  (void)ws_size; (void)n_in; (void)in_sizes;

  const int TB = 256;
  auto cdiv = [](long long a, long long b) { return (int)((a + b - 1) / b); };

  // ---- weight repack / convert -----------------------------------------
  auto repack = [&](const float* s, half_t* d, int Cin, int CinPad, int Cout) {
    int n = 9 * Cout * CinPad;
    k_repack_w<<<cdiv(n, TB), TB, 0, stream>>>(s, d, Cin, CinPad, Cout);
  };
  repack(co_w0, w0_16,  CIN0,   CIN0P,  Cc);
  repack(co_w1, w1_16,  Cc,     Cc,     Cc);
  repack(co_w2, w2_16,  Cc,     Cc,     Cc);
  repack(co_w3, w3_16,  Cc,     Cc,     COFF);
  repack(bb_w0, wbb16,  3 * Cc, 3 * Cc, Cc);
  repack(r1w1,  wr1a16, Cc,     Cc,     Cc);
  repack(r1w2,  wr1b16, Cc,     Cc,     Cc);
  repack(r2w1,  wr2a16, Cc,     Cc,     Cc);
  repack(r2w2,  wr2b16, Cc,     Cc,     Cc);
  k_f32_to_f16<<<cdiv(Cc * KKD, TB), TB, 0, stream>>>(dc_w, wdc16, Cc * KKD);
  k_zero16<<<1, 256, 0, stream>>>(zrow, CIN0P);

  // ---- warp + concat (NHWC) --------------------------------------------
  k_prep_xcat<<<cdiv(P, TB), TB, 0, stream>>>(feat_prop, grid_n1, feat_n2, grid_n2,
                                              feat_cur, flow_1, flow_2, xcat16);

  // ---- conv helper: 8 waves per block share one cout-tile ---------------
  auto conv = [&](const half_t* in16, int CinPad, const half_t* wT, const float* bias,
                  int Cout, const float* res, float* o32, half_t* o16,
                  int c16tot, int c16base, int act) {
    int blocks = (Cout / 16) * ((P / 16) / WPB);     // exact grid, no guards
    size_t shmem = (size_t)9 * 16 * CinPad * 2;      // <= 64512 B (conv0)
    k_conv3x3_wmma<<<blocks, TB, shmem, stream>>>(
        in16, CinPad, wT, zrow, bias, Cout, res, o32, o16, c16tot, c16base, act);
  };

  // ---- offset head ------------------------------------------------------
  conv(xcat16, CIN0P, w0_16, co_b0, Cc,   nullptr, nullptr, hA16, Cc, 0, 1);
  conv(hA16,   Cc,    w1_16, co_b1, Cc,   nullptr, nullptr, hB16, Cc, 0, 1);
  conv(hB16,   Cc,    w2_16, co_b2, Cc,   nullptr, nullptr, hA16, Cc, 0, 1);
  conv(hA16,   Cc,    w3_16, co_b3, COFF, nullptr, co3o32, nullptr, 0, 0, 0);

  k_offset_mask<<<cdiv((long long)Bn * COFF * HW, TB), TB, 0, stream>>>(
      co3o32, flow_1, flow_2, off32, msk32);

  // ---- deformable sampling + GEMM --------------------------------------
  k_deform_val<<<cdiv((long long)Bn * HW * DG * KT, TB), TB, 0, stream>>>(
      feat_prop, feat_n2, off32, msk32, val16);
  {
    int blocks = (Cc / 16) * ((P / 16) / WPB);       // 576
    size_t shmem = (size_t)16 * KKD * 2;             // 36864 B
    k_dgemm_wmma<<<blocks, TB, shmem, stream>>>(
        val16, wdc16, dc_b, fpn32, fcat16, 3 * Cc, 2 * Cc);
  }

  // ---- backbone ---------------------------------------------------------
  k_prefix16<<<cdiv((long long)Bn * 2 * Cc * HW, TB), TB, 0, stream>>>(bb_prefix, fcat16);

  conv(fcat16, 3 * Cc, wbb16,  bb_b0, Cc, nullptr, h0_32, h0_16, Cc, 0, 1);   // lrelu
  conv(h0_16,  Cc,     wr1a16, r1b1,  Cc, nullptr, nullptr, rA16, Cc, 0, 2);  // relu
  conv(rA16,   Cc,     wr1b16, r1b2,  Cc, h0_32,   h1_32,  h1_16, Cc, 0, 0);  // + res
  conv(h1_16,  Cc,     wr2a16, r2b1,  Cc, nullptr, nullptr, rB16, Cc, 0, 2);  // relu
  conv(rB16,   Cc,     wr2b16, r2b2,  Cc, h1_32,   h2_32,  nullptr, 0, 0, 0); // + res

  // ---- final: feat_prop_new + h ----------------------------------------
  k_final_add<<<cdiv((long long)P * Cc, TB), TB, 0, stream>>>(
      fpn32, h2_32, (float*)d_out, P * Cc);
}
